// JK_4913442586831
// MI455X (gfx1250) — matrix operand order repaired and checked by reading, not compiled, not run on hardware
//
#include <hip/hip_runtime.h>

typedef __attribute__((ext_vector_type(16))) _Float16 v16h;
typedef __attribute__((ext_vector_type(8)))  _Float16 v8h;
typedef __attribute__((ext_vector_type(8)))  float    v8f;

#define TPB 256
#define KDIM 128   // all GEMMs in this model have K == 128 (compile-time unroll)

// ---------------------------------------------------------------- elementwise
__global__ void k_cvt_f32_f16(const float* __restrict__ in,
                              _Float16* __restrict__ out, int n) {
  int i = blockIdx.x * blockDim.x + threadIdx.x;
  if (i < n) out[i] = (_Float16)in[i];
}

// Bt[n][k] = (n < Nvalid) ? W[k][n] : 0   -- f16, K-contiguous rows, N padded
__global__ void k_transpose_w(const float* __restrict__ W, _Float16* __restrict__ Bt,
                              int K, int Nvalid, int Npad) {
  int i = blockIdx.x * blockDim.x + threadIdx.x;
  if (i < Npad * K) {
    int n = i / K, k = i % K;
    Bt[i] = (n < Nvalid) ? (_Float16)W[(size_t)k * Nvalid + n] : (_Float16)0.0f;
  }
}

__global__ void k_zero_i32(int* __restrict__ p, int n) {
  int i = blockIdx.x * blockDim.x + threadIdx.x;
  if (i < n) p[i] = 0;
}

__global__ void k_degree(const int* __restrict__ dst, int* __restrict__ deg, int E) {
  int e = blockIdx.x * blockDim.x + threadIdx.x;
  if (e < E) atomicAdd(&deg[dst[e]], 1);
}

__global__ void k_dinv(const int* __restrict__ deg, float* __restrict__ dinv, int n) {
  int i = blockIdx.x * blockDim.x + threadIdx.x;
  if (i < n) dinv[i] = rsqrtf((float)deg[i] + 1.0f);  // self loop included
}

// agg[i] = h[i] * dinv[node]^2   (self-loop term, also initializes agg)
__global__ void k_agg_init(const float* __restrict__ h, const float* __restrict__ dinv,
                           float* __restrict__ agg, int total, int nh) {
  int i = blockIdx.x * blockDim.x + threadIdx.x;
  if (i < total) {
    float d = dinv[i / nh];
    agg[i] = h[i] * d * d;
  }
}

// one wave32 per edge; nh == 128 -> 4 floats per lane.
// e is wave-uniform: force it scalar so src/dst/dinv become s_load broadcasts.
__global__ void k_agg_edges(const int* __restrict__ src, const int* __restrict__ dst,
                            const float* __restrict__ dinv, const float* __restrict__ h,
                            float* __restrict__ agg, int E) {
  int t = blockIdx.x * blockDim.x + threadIdx.x;
  int e = __builtin_amdgcn_readfirstlane(t >> 5);   // uniform edge id -> SGPR
  int lane = t & 31;
  if (e >= E) return;                               // wave-uniform exit
  int s = src[e], d = dst[e];
  float norm = dinv[s] * dinv[d];
  const float4 v = ((const float4*)(h + (size_t)s * 128))[lane];
  float* ad = agg + (size_t)d * 128 + lane * 4;
  unsafeAtomicAdd(ad + 0, v.x * norm);  // global_atomic_add_f32, L2-resident
  unsafeAtomicAdd(ad + 1, v.y * norm);
  unsafeAtomicAdd(ad + 2, v.z * norm);
  unsafeAtomicAdd(ad + 3, v.w * norm);
}

// h1 = relu(agg + b); also emit f16 copy for the next WMMA GEMM
__global__ void k_finalize1(const float* __restrict__ agg, const float* __restrict__ bias,
                            float* __restrict__ h1, _Float16* __restrict__ h1h,
                            int total, int nh) {
  int i = blockIdx.x * blockDim.x + threadIdx.x;
  if (i < total) {
    float v = agg[i] + bias[i % nh];
    v = v > 0.0f ? v : 0.0f;
    h1[i] = v;
    h1h[i] = (_Float16)v;
  }
}

// hjk = max(h1, relu(agg + b)) -> f16 for final GEMM
__global__ void k_finalize2(const float* __restrict__ agg, const float* __restrict__ bias,
                            const float* __restrict__ h1, _Float16* __restrict__ hjkh,
                            int total, int nh) {
  int i = blockIdx.x * blockDim.x + threadIdx.x;
  if (i < total) {
    float v = agg[i] + bias[i % nh];
    v = v > 0.0f ? v : 0.0f;
    hjkh[i] = (_Float16)fmaxf(h1[i], v);
  }
}

// ---------------------------------------------------------------- WMMA GEMM
// C[M x Nvalid] = A[M x 128](f16) @ B[128 x Nvalid](f16) (+ bias), f32 acc.
// B is pre-transposed/padded: Bt[NTILES*16][128] with Bt[n][k] = B[k][n].
// Each wave computes one 16-row strip x NTILES 16-col tiles, reusing the A
// fragment across NTILES accumulators. K-loop fully unrolled ->
// 4 x NTILES independent v_wmma_f32_16x16x32_f16 per wave.
// Tile selection is wave-uniform so EXEC is all-1s at every WMMA.
template <int NTILES>
__global__ void k_wmma_gemm(const _Float16* __restrict__ A,
                            const _Float16* __restrict__ Bt,
                            float* __restrict__ C, int ldc,
                            const float* __restrict__ bias,
                            int M, int Nvalid) {
  const int wavesPerBlock = blockDim.x >> 5;
  int mt = blockIdx.x * wavesPerBlock + (threadIdx.x >> 5);
  int Mtiles = (M + 15) >> 4;
  if (mt >= Mtiles) return;                    // wave-uniform exit

  int lane = threadIdx.x & 31;
  int hs = lane >> 4;       // lane half-select (0: lanes 0-15, 1: lanes 16-31)
  int mr = lane & 15;

  int arow = mt * 16 + mr;
  const _Float16* Arow = A + (size_t)(arow < M ? arow : 0) * KDIM;

  v8f acc[NTILES];
#pragma unroll
  for (int t = 0; t < NTILES; ++t) acc[t] = (v8f){};

#pragma unroll
  for (int k0 = 0; k0 < KDIM; k0 += 32) {
    // A fragment (16x32 f16): element i -> K = k0 + 16*hs + i; one 32B vector load.
    v16h a = *(const v16h*)(Arow + k0 + 16 * hs);
#pragma unroll
    for (int t = 0; t < NTILES; ++t) {
      // B fragment (32x16 f16): element i is K = k0 + 16*hs + i, N = t*16+mr.
      // In Bt this is one contiguous, 32B-aligned run of 16 halves.
      v16h b = *(const v16h*)(Bt + (size_t)(t * 16 + mr) * KDIM + k0 + 16 * hs);
      acc[t] = __builtin_amdgcn_wmma_f32_16x16x32_f16(
          /*neg_a=*/false, a, /*neg_b=*/false, b,
          /*c_mod=*/(short)0, acc[t], /*reuse_a=*/false, /*reuse_b=*/false);
    }
  }

  // Epilogue: fullM is wave-uniform (always true for M % 16 == 0);
  // the N-edge test per tile is an SGPR compare -> uniform branch, so the
  // per-lane guard only materializes on the single partial classifier tile.
  const bool fullM = (mt * 16 + 16) <= M;
#pragma unroll
  for (int t = 0; t < NTILES; ++t) {
    const int ncol = t * 16 + mr;
    const bool tileFullN = ((t + 1) * 16) <= Nvalid;   // wave-uniform
    if (!tileFullN && ncol >= Nvalid) continue;        // lane guard, edge tile only
    const float bv = bias ? bias[ncol] : 0.0f;
    float* Cp = C + (size_t)(mt * 16 + 8 * hs) * ldc + ncol;
    if (fullM) {
#pragma unroll
      for (int v = 0; v < 8; ++v)
        Cp[(size_t)v * ldc] = acc[t][v] + bv;          // unguarded, coalesced
    } else {
#pragma unroll
      for (int v = 0; v < 8; ++v)
        if (mt * 16 + 8 * hs + v < M) Cp[(size_t)v * ldc] = acc[t][v] + bv;
    }
  }
}

// ---------------------------------------------------------------- launch
static inline int cdiv(int a, int b) { return (a + b - 1) / b; }

// Only the two shapes this model uses are instantiated: NTILES=8 (128-wide
// hidden layers) and NTILES=3 (40->48-padded classifier).
static void launch_gemm(int ntiles, const _Float16* A, const _Float16* Bt, float* C,
                        int ldc, const float* bias, int M, int Nvalid,
                        hipStream_t stream) {
  const int WPB = TPB / 32;
  int blocks = cdiv(cdiv(M, 16), WPB);
  if (ntiles == 3)
    k_wmma_gemm<3><<<blocks, TPB, 0, stream>>>(A, Bt, C, ldc, bias, M, Nvalid);
  else
    k_wmma_gemm<8><<<blocks, TPB, 0, stream>>>(A, Bt, C, ldc, bias, M, Nvalid);
}

extern "C" void kernel_launch(void* const* d_in, const int* in_sizes, int n_in,
                              void* d_out, int out_size, void* d_ws, size_t ws_size,
                              hipStream_t stream) {
  const float* x   = (const float*)d_in[0];
  const int*   ei  = (const int*)d_in[1];
  const float* W1  = (const float*)d_in[2];
  const float* b1  = (const float*)d_in[3];
  const float* Wx  = (const float*)d_in[4];
  const float* bx  = (const float*)d_in[5];
  const float* Wfc = (const float*)d_in[6];
  const float* bfc = (const float*)d_in[7];
  float* out = (float*)d_out;

  const int nhid   = in_sizes[3];              // 128
  const int nfeat  = in_sizes[2] / nhid;       // 128 (== KDIM)
  const int nclass = in_sizes[7];              // 40
  const int N      = in_sizes[0] / nfeat;      // 50000
  const int E      = in_sizes[1] / 2;          // 800000
  const int* src = ei;
  const int* dst = ei + E;

  const int ntilesH = cdiv(nhid, 16);          // 8
  const int ntilesC = cdiv(nclass, 16);        // 3
  const int npadH = ntilesH * 16;              // 128
  const int npadC = ntilesC * 16;              // 48

  // ---- carve workspace (256B aligned slices)
  char* w = (char*)d_ws;
  auto carve = [&](size_t bytes) -> char* {
    char* p = w;
    w += (bytes + 255) & ~(size_t)255;
    return p;
  };
  int*      deg   = (int*)carve((size_t)N * 4);
  float*    dinv  = (float*)carve((size_t)N * 4);
  _Float16* xh    = (_Float16*)carve((size_t)N * nfeat * 2);
  _Float16* h1h   = (_Float16*)carve((size_t)N * nhid * 2);
  _Float16* hjkh  = (_Float16*)carve((size_t)N * nhid * 2);
  float*    h     = (float*)carve((size_t)N * nhid * 4);
  float*    agg   = (float*)carve((size_t)N * nhid * 4);
  float*    h1    = (float*)carve((size_t)N * nhid * 4);
  _Float16* w1t   = (_Float16*)carve((size_t)npadH * KDIM * 2);
  _Float16* wxt   = (_Float16*)carve((size_t)npadH * KDIM * 2);
  _Float16* wfct  = (_Float16*)carve((size_t)npadC * KDIM * 2);
  (void)ws_size;

  const int total = N * nhid;

  // ---- f16 input copy + transposed/padded f16 weights
  k_cvt_f32_f16<<<cdiv(N * nfeat, TPB), TPB, 0, stream>>>(x, xh, N * nfeat);
  k_transpose_w<<<cdiv(npadH * KDIM, TPB), TPB, 0, stream>>>(W1, w1t, KDIM, nhid, npadH);
  k_transpose_w<<<cdiv(npadH * KDIM, TPB), TPB, 0, stream>>>(Wx, wxt, KDIM, nhid, npadH);
  k_transpose_w<<<cdiv(npadC * KDIM, TPB), TPB, 0, stream>>>(Wfc, wfct, KDIM, nclass, npadC);

  // ---- degrees & symmetric norm
  k_zero_i32<<<cdiv(N, TPB), TPB, 0, stream>>>(deg, N);
  k_degree<<<cdiv(E, TPB), TPB, 0, stream>>>(dst, deg, E);
  k_dinv<<<cdiv(N, TPB), TPB, 0, stream>>>(deg, dinv, N);

  // ---- layer 1: h = x @ W1 ; aggregate ; relu
  launch_gemm(ntilesH, xh, w1t, h, nhid, nullptr, N, nhid, stream);
  k_agg_init<<<cdiv(total, TPB), TPB, 0, stream>>>(h, dinv, agg, total, nhid);
  k_agg_edges<<<cdiv(E * 32, TPB), TPB, 0, stream>>>(src, dst, dinv, h, agg, E);
  k_finalize1<<<cdiv(total, TPB), TPB, 0, stream>>>(agg, b1, h1, h1h, total, nhid);

  // ---- layer 2: h = h1 @ Wx ; aggregate ; relu ; JK max
  launch_gemm(ntilesH, h1h, wxt, h, nhid, nullptr, N, nhid, stream);
  k_agg_init<<<cdiv(total, TPB), TPB, 0, stream>>>(h, dinv, agg, total, nhid);
  k_agg_edges<<<cdiv(E * 32, TPB), TPB, 0, stream>>>(src, dst, dinv, h, agg, E);
  k_finalize2<<<cdiv(total, TPB), TPB, 0, stream>>>(agg, bx, h1, hjkh, total, nhid);

  // ---- classifier: out = hjk @ Wfc + bfc  (40 padded to 48, guarded store)
  launch_gemm(ntilesC, hjkh, wfct, out, nclass, bfc, N, nclass, stream);
  (void)out_size;
  (void)n_in;
}
